// PublicCausalAdaptiveKernelAttention_16956530885215
// MI455X (gfx1250) — matrix operand order repaired and checked by reading, not compiled, not run on hardware
//
#include <hip/hip_runtime.h>
#include <math.h>

// ---------------- types ----------------
typedef __attribute__((ext_vector_type(16))) __bf16 v16bf;
typedef __attribute__((ext_vector_type(8)))  __bf16 v8bf;
typedef __attribute__((ext_vector_type(8)))  float  v8f;
typedef __attribute__((ext_vector_type(4)))  int    v4i_;

#define GAS_ __attribute__((address_space(1)))
#define LAS_ __attribute__((address_space(3)))

#define BS_   4
#define T_    1024
#define D_    512
#define H_    8
#define HD_   64
#define TH_   256
#define CTX_  1024

// CDNA5 async global->LDS path (ASYNCcnt), with compile-safe fallback
#if defined(__has_builtin)
#if __has_builtin(__builtin_amdgcn_global_load_async_to_lds_b128) && \
    __has_builtin(__builtin_amdgcn_s_wait_asynccnt)
#define USE_ASYNC_LDS 1
#endif
#endif
#ifndef USE_ASYNC_LDS
#define USE_ASYNC_LDS 0
#endif

// ---------------- helpers ----------------
__device__ __forceinline__ float gelu_exact(float x) {
    return 0.5f * x * (1.0f + erff(x * 0.70710678118654752440f));
}

__device__ __forceinline__ float blockReduceSum(float v, float* red) {
    #pragma unroll
    for (int off = 16; off > 0; off >>= 1) v += __shfl_xor(v, off, 32);
    int wave = threadIdx.x >> 5, lane = threadIdx.x & 31;
    if (lane == 0) red[wave] = v;
    __syncthreads();
    int nw = (blockDim.x + 31) >> 5;
    float t = (threadIdx.x < (unsigned)nw) ? red[threadIdx.x] : 0.0f;
    if (wave == 0) {
        #pragma unroll
        for (int off = 4; off > 0; off >>= 1) t += __shfl_xor(t, off, 32);
        if (lane == 0) red[0] = t;
    }
    __syncthreads();
    float r = red[0];
    __syncthreads();
    return r;
}

__device__ __forceinline__ float blockReduceMax(float v, float* red) {
    #pragma unroll
    for (int off = 16; off > 0; off >>= 1) v = fmaxf(v, __shfl_xor(v, off, 32));
    int wave = threadIdx.x >> 5, lane = threadIdx.x & 31;
    if (lane == 0) red[wave] = v;
    __syncthreads();
    int nw = (blockDim.x + 31) >> 5;
    float t = (threadIdx.x < (unsigned)nw) ? red[threadIdx.x] : -3.0e38f;
    if (wave == 0) {
        #pragma unroll
        for (int off = 4; off > 0; off >>= 1) t = fmaxf(t, __shfl_xor(t, off, 32));
        if (lane == 0) red[0] = t;
    }
    __syncthreads();
    float r = red[0];
    __syncthreads();
    return r;
}

// ---------------- fp32 -> bf16 convert ----------------
__global__ __launch_bounds__(256) void f2bf_kernel(const float* __restrict__ s,
                                                   __bf16* __restrict__ d, long n) {
    long i = (long)blockIdx.x * 256 + threadIdx.x;
    if (i < n) d[i] = (__bf16)s[i];
}

// ---------------- generic batched NT WMMA GEMM ----------------
// C[z][m][n] = scale * sum_k A[z][m][k] * B[z][n][k]  (+ bias[n])
// A,B bf16 row-major; C fp32 or bf16. Per-z offsets: (z/Hdiv)*sB + (z%Hdiv)*sH.
// Block: nwaves = blockDim/32 waves; each wave one 16x16 C tile (16 x nwaves*16).
// Whole 16xK A panel (K <= 1024) is staged in LDS ONCE (async DMA when available),
// so the K loop is barrier-free: 2x ds_load_b128 + 2x global_load_b128 + v_wmma.
#define KMAX_ 1024
#define APAD_ (KMAX_ + 8)   // +16B/row: stride 2064B == 16 mod 256 -> conflict-free
__global__ __launch_bounds__(256) void gemm_nt_bf16(
    const __bf16* __restrict__ A, const __bf16* __restrict__ B,
    float* __restrict__ Cf, __bf16* __restrict__ Cb,
    const float* __restrict__ bias,
    int M, int N, int K, int lda, int ldb, int ldc,
    long aB, long aH, long bB, long bH, long cB, long cH, int Hdiv,
    float scale)
{
    __shared__ __bf16 As[16][APAD_];

    int z  = blockIdx.z;
    int zb = z / Hdiv, zh = z % Hdiv;
    const __bf16* Ap = A + zb * aB + zh * aH;
    const __bf16* Bp = B + zb * bB + zh * bH;
    long coff = zb * cB + zh * cH;

    int m0     = blockIdx.y * 16;
    int wave   = threadIdx.x >> 5;
    int lane   = threadIdx.x & 31;
    int nwaves = blockDim.x >> 5;
    int n0     = blockIdx.x * (nwaves * 16) + wave * 16;
    int ln = lane & 15;
    int lh = lane >> 4;

    // ---- stage full 16 x K A panel into LDS (128-bit granules) ----
    {
        int cpr   = K >> 3;          // b128 chunks per row
        int total = cpr << 4;        // 16 rows
        for (int t = threadIdx.x; t < total; t += blockDim.x) {
            int r = t / cpr;
            int c = (t - r * cpr) << 3;
            const __bf16* src = Ap + (long)(m0 + r) * lda + c;
#if USE_ASYNC_LDS
            __builtin_amdgcn_global_load_async_to_lds_b128(
                (GAS_ v4i_*)src, (LAS_ v4i_*)&As[r][c], 0, 0);
#else
            *(v8bf*)&As[r][c] = *(const v8bf*)src;
#endif
        }
#if USE_ASYNC_LDS
        __builtin_amdgcn_s_wait_asynccnt(0);
#endif
        __syncthreads();
    }

    if (n0 < N) {
        v8f acc = {};
        const __bf16* brow = Bp + (long)(n0 + ln) * ldb + lh * 16;

        #pragma unroll 2
        for (int k0 = 0; k0 < K; k0 += 32) {
            // A fragment: lanes 0-15: K 0-15; lanes 16-31: K 16-31 (matches B)
            int abase = k0 + lh * 16;
            v8bf alo = *(const v8bf*)&As[ln][abase];
            v8bf ahi = *(const v8bf*)&As[ln][abase + 8];
            v16bf a = __builtin_shufflevector(alo, ahi,
                0, 1, 2, 3, 4, 5, 6, 7, 8, 9, 10, 11, 12, 13, 14, 15);

            // B fragment: lane holds 16 contiguous K for column n0+ln
            v16bf b = *(const v16bf*)(brow + k0);
            __builtin_prefetch((const void*)(brow + k0 + 64), 0, 0);

            acc = __builtin_amdgcn_wmma_f32_16x16x32_bf16(
                /*neg_a=*/false, a, /*neg_b=*/false, b,
                /*c_mod=*/(short)0, acc, /*reuse_a=*/false, /*reuse_b=*/false);
        }

        int col = n0 + ln;
        float bv = bias ? bias[col] : 0.0f;
        #pragma unroll
        for (int r = 0; r < 8; ++r) {
            int row = m0 + lh * 8 + r;
            float v = acc[r] * scale + bv;
            long idx = coff + (long)row * ldc + col;
            if (Cf) Cf[idx] = v;
            else    Cb[idx] = (__bf16)v;
        }
    }
}

// ---------------- history embed: eh = gelu(LN(hist @ hm_W.T + hm_b)) ----------------
__global__ __launch_bounds__(256) void hist_embed_kernel(
    const float* __restrict__ hist, const float* __restrict__ W,
    const float* __restrict__ b, const float* __restrict__ g,
    const float* __restrict__ beta, float* __restrict__ eh)
{
    int row = blockIdx.x;  // bs*TH + th
    __shared__ float hs[D_];
    __shared__ float red[8];
    for (int k = threadIdx.x; k < D_; k += 256) hs[k] = hist[(long)row * D_ + k];
    __syncthreads();

    float v[4];
    float s = 0.0f, s2 = 0.0f;
    #pragma unroll
    for (int q = 0; q < 4; ++q) {
        int oc = threadIdx.x + q * 256;
        const float* wr = W + (long)oc * D_;
        float acc = b[oc];
        #pragma unroll 8
        for (int k = 0; k < D_; ++k) acc += hs[k] * wr[k];
        v[q] = acc; s += acc; s2 += acc * acc;
    }
    float S  = blockReduceSum(s, red);
    float S2 = blockReduceSum(s2, red);
    float mu  = S * (1.0f / CTX_);
    float var = S2 * (1.0f / CTX_) - mu * mu;
    float inv = rsqrtf(var + 1e-5f);
    #pragma unroll
    for (int q = 0; q < 4; ++q) {
        int oc = threadIdx.x + q * 256;
        float y = (v[q] - mu) * inv * g[oc] + beta[oc];
        eh[(long)row * CTX_ + oc] = gelu_exact(y);
    }
}

// ---------------- attention-pool scores: awl = gelu(eh@ca_W1.T+b1)@ca_W2.T+b2 ----------------
__global__ __launch_bounds__(256) void ca_score_kernel(
    const float* __restrict__ eh, const float* __restrict__ W1,
    const float* __restrict__ b1, const float* __restrict__ W2,
    const float* __restrict__ b2, float* __restrict__ awl)
{
    int row = blockIdx.x;  // bs*TH + th
    __shared__ float es[CTX_];
    __shared__ float red[8];
    for (int k = threadIdx.x; k < CTX_; k += 256) es[k] = eh[(long)row * CTX_ + k];
    __syncthreads();

    int j = threadIdx.x;  // CTX/4 == 256 outputs
    const float* wr = W1 + (long)j * CTX_;
    float acc = b1[j];
    #pragma unroll 8
    for (int k = 0; k < CTX_; ++k) acc += es[k] * wr[k];
    float part = gelu_exact(acc) * W2[j];
    float tot = blockReduceSum(part, red);
    if (threadIdx.x == 0) awl[row] = tot + b2[0];
}

// ---------------- softmax over TH + weighted pool -> ctx ----------------
__global__ __launch_bounds__(256) void ctx_pool_kernel(
    const float* __restrict__ eh, const float* __restrict__ awl,
    float* __restrict__ ctx)
{
    int bs = blockIdx.x;
    __shared__ float wgt[TH_];
    __shared__ float red[8];
    float a = awl[bs * TH_ + threadIdx.x];
    float mx = blockReduceMax(a, red);
    float e = expf(a - mx);
    float sm = blockReduceSum(e, red);
    wgt[threadIdx.x] = e / sm;
    __syncthreads();

    #pragma unroll
    for (int q = 0; q < 4; ++q) {
        int c = threadIdx.x + q * 256;
        float acc = 0.0f;
        #pragma unroll 4
        for (int th = 0; th < TH_; ++th)
            acc += eh[((long)bs * TH_ + th) * CTX_ + c] * wgt[th];
        ctx[(long)bs * CTX_ + c] = acc;
    }
}

// ---------------- adaptive kernel generator ----------------
__global__ __launch_bounds__(256) void kernel_gen_kernel(
    const float* __restrict__ ctx, const float* __restrict__ W1,
    const float* __restrict__ b1, const float* __restrict__ g,
    const float* __restrict__ beta, const float* __restrict__ W2,
    const float* __restrict__ b2, float* __restrict__ kern)
{
    int bs = blockIdx.x;
    __shared__ float cs[CTX_];
    __shared__ float t1[D_];
    __shared__ float kp[72];
    __shared__ float red[8];
    for (int k = threadIdx.x; k < CTX_; k += 256) cs[k] = ctx[(long)bs * CTX_ + k];
    __syncthreads();

    float v[2];
    float s = 0.0f, s2 = 0.0f;
    #pragma unroll
    for (int q = 0; q < 2; ++q) {
        int oc = threadIdx.x + q * 256;
        const float* wr = W1 + (long)oc * CTX_;
        float acc = b1[oc];
        #pragma unroll 8
        for (int k = 0; k < CTX_; ++k) acc += cs[k] * wr[k];
        v[q] = acc; s += acc; s2 += acc * acc;
    }
    float S  = blockReduceSum(s, red);
    float S2 = blockReduceSum(s2, red);
    float mu  = S * (1.0f / D_);
    float var = S2 * (1.0f / D_) - mu * mu;
    float inv = rsqrtf(var + 1e-5f);
    #pragma unroll
    for (int q = 0; q < 2; ++q) {
        int oc = threadIdx.x + q * 256;
        t1[oc] = gelu_exact((v[q] - mu) * inv * g[oc] + beta[oc]);
    }
    __syncthreads();

    if (threadIdx.x < 72) {
        const float* wr = W2 + (long)threadIdx.x * D_;
        float acc = b2[threadIdx.x];
        #pragma unroll 8
        for (int k = 0; k < D_; ++k) acc += t1[k] * wr[k];
        kp[threadIdx.x] = acc;
    }
    __syncthreads();

    if (threadIdx.x < H_) {
        int h = threadIdx.x;
        float mx = -3.0e38f;
        #pragma unroll
        for (int j = 0; j < 9; ++j) mx = fmaxf(mx, kp[h * 9 + j]);
        float sm = 0.0f;
        float e[9];
        #pragma unroll
        for (int j = 0; j < 9; ++j) { e[j] = expf(kp[h * 9 + j] - mx); sm += e[j]; }
        #pragma unroll
        for (int j = 0; j < 9; ++j) kern[(long)(bs * H_ + h) * 9 + j] = e[j] / sm;
    }
}

// ---------------- depthwise 3x3 conv + causal mask + row softmax -> bf16 attn ----------------
__global__ __launch_bounds__(256) void conv_mask_softmax_kernel(
    const float* __restrict__ scores, const float* __restrict__ kern,
    __bf16* __restrict__ attn)
{
    int i = blockIdx.x;   // query row
    int z = blockIdx.y;   // bs*H + h
    __shared__ float Ls[3][T_ + 2];
    __shared__ float red[8];

    const float* w = kern + (long)z * 9;
    float w00 = w[0], w01 = w[1], w02 = w[2];
    float w10 = w[3], w11 = w[4], w12 = w[5];
    float w20 = w[6], w21 = w[7], w22 = w[8];

    const float* sp = scores + (long)z * T_ * T_;
    #pragma unroll
    for (int rr = 0; rr < 3; ++rr) {
        int r = i + rr - 1;
        for (int j = threadIdx.x; j < T_ + 2; j += 256) {
            int c = j - 1;
            float v = 0.0f;
            if (r >= 0 && r < T_ && c >= 0 && c < T_ && c <= r)
                v = sp[(long)r * T_ + c];
            Ls[rr][j] = v;
        }
    }
    __syncthreads();

    float vals[4];
    float mymax = -3.0e38f;
    #pragma unroll
    for (int q = 0; q < 4; ++q) {
        int j = threadIdx.x + q * 256;
        float c = w00 * Ls[0][j] + w01 * Ls[0][j + 1] + w02 * Ls[0][j + 2]
                + w10 * Ls[1][j] + w11 * Ls[1][j + 1] + w12 * Ls[1][j + 2]
                + w20 * Ls[2][j] + w21 * Ls[2][j + 1] + w22 * Ls[2][j + 2];
        if (j > i) c = -1.0e9f;
        vals[q] = c;
        mymax = fmaxf(mymax, c);
    }
    float gmax = blockReduceMax(mymax, red);
    float mysum = 0.0f;
    #pragma unroll
    for (int q = 0; q < 4; ++q) {
        vals[q] = expf(vals[q] - gmax);
        mysum += vals[q];
    }
    float gsum = blockReduceSum(mysum, red);
    float rinv = 1.0f / gsum;
    #pragma unroll
    for (int q = 0; q < 4; ++q) {
        int j = threadIdx.x + q * 256;
        attn[((long)z * T_ + i) * T_ + j] = (__bf16)(vals[q] * rinv);
    }
}

// ---------------- V transpose: [BS,T,D] -> [BS,H,HD,T] (bf16) ----------------
__global__ __launch_bounds__(256) void transpose_v_kernel(
    const __bf16* __restrict__ v, __bf16* __restrict__ vt)
{
    long idx = (long)blockIdx.x * 256 + threadIdx.x;  // BS*H*HD*T
    if (idx >= (long)BS_ * H_ * HD_ * T_) return;
    int t = (int)(idx & (T_ - 1));
    long r = idx >> 10;
    int d = (int)(r & (HD_ - 1)); r >>= 6;
    int h = (int)(r & (H_ - 1));
    int bs = (int)(r >> 3);
    vt[idx] = v[((long)bs * T_ + t) * D_ + h * HD_ + d];
}

// ---------------- head merge: [BS,H,T,HD] -> [BS,T,D] (bf16) ----------------
__global__ __launch_bounds__(256) void merge_heads_kernel(
    const __bf16* __restrict__ ao, __bf16* __restrict__ aom)
{
    long idx = (long)blockIdx.x * 256 + threadIdx.x;  // BS*T*D
    if (idx >= (long)BS_ * T_ * D_) return;
    int c = (int)(idx & (D_ - 1));
    long r = idx >> 9;
    int t = (int)(r & (T_ - 1));
    int bs = (int)(r >> 10);
    int h = c >> 6, d = c & (HD_ - 1);
    aom[idx] = ao[(((long)bs * H_ + h) * T_ + t) * HD_ + d];
}

// ---------------- launcher ----------------
extern "C" void kernel_launch(void* const* d_in, const int* in_sizes, int n_in,
                              void* d_out, int out_size, void* d_ws, size_t ws_size,
                              hipStream_t stream) {
    (void)in_sizes; (void)n_in; (void)out_size; (void)ws_size;

    const float* x       = (const float*)d_in[0];
    const float* hist    = (const float*)d_in[4];
    const float* Wq      = (const float*)d_in[5];
    const float* Wk      = (const float*)d_in[6];
    const float* Wv      = (const float*)d_in[7];
    const float* Wo      = (const float*)d_in[8];
    const float* bo      = (const float*)d_in[9];
    const float* hm_W    = (const float*)d_in[10];
    const float* hm_b    = (const float*)d_in[11];
    const float* hm_g    = (const float*)d_in[12];
    const float* hm_beta = (const float*)d_in[13];
    const float* ca_W1   = (const float*)d_in[14];
    const float* ca_b1   = (const float*)d_in[15];
    const float* ca_W2   = (const float*)d_in[16];
    const float* ca_b2   = (const float*)d_in[17];
    const float* kg_W1   = (const float*)d_in[18];
    const float* kg_b1   = (const float*)d_in[19];
    const float* kg_g    = (const float*)d_in[20];
    const float* kg_beta = (const float*)d_in[21];
    const float* kg_W2   = (const float*)d_in[22];
    const float* kg_b2   = (const float*)d_in[23];

    // workspace carve-up (256B aligned)
    char* ws = (char*)d_ws;
    size_t off = 0;
    auto take = [&](size_t bytes) -> char* {
        char* p = ws + off;
        off = (off + bytes + 255) & ~(size_t)255;
        return p;
    };
    const long MBT  = (long)BS_ * T_;              // 4096
    __bf16* xb   = (__bf16*)take(MBT * D_ * 2);
    __bf16* wqb  = (__bf16*)take((long)D_ * D_ * 2);
    __bf16* wkb  = (__bf16*)take((long)D_ * D_ * 2);
    __bf16* wvb  = (__bf16*)take((long)D_ * D_ * 2);
    __bf16* wob  = (__bf16*)take((long)D_ * D_ * 2);
    __bf16* qb   = (__bf16*)take(MBT * D_ * 2);
    __bf16* kb   = (__bf16*)take(MBT * D_ * 2);
    __bf16* vb   = (__bf16*)take(MBT * D_ * 2);
    float*  scb  = (float*) take((long)BS_ * H_ * T_ * T_ * 4);   // 128 MiB
    __bf16* atb  = (__bf16*)take((long)BS_ * H_ * T_ * T_ * 2);   //  64 MiB
    __bf16* vtb  = (__bf16*)take(MBT * D_ * 2);
    __bf16* aob  = (__bf16*)take(MBT * D_ * 2);
    __bf16* aomb = (__bf16*)take(MBT * D_ * 2);
    float*  ehb  = (float*) take((long)BS_ * TH_ * CTX_ * 4);
    float*  awlb = (float*) take((long)BS_ * TH_ * 4);
    float*  ctxb = (float*) take((long)BS_ * CTX_ * 4);
    float*  krnb = (float*) take((long)BS_ * H_ * 9 * 4);

    // 1) bf16 conversions
    f2bf_kernel<<<(MBT * D_ + 255) / 256, 256, 0, stream>>>(x, xb, MBT * D_);
    f2bf_kernel<<<((long)D_ * D_ + 255) / 256, 256, 0, stream>>>(Wq, wqb, (long)D_ * D_);
    f2bf_kernel<<<((long)D_ * D_ + 255) / 256, 256, 0, stream>>>(Wk, wkb, (long)D_ * D_);
    f2bf_kernel<<<((long)D_ * D_ + 255) / 256, 256, 0, stream>>>(Wv, wvb, (long)D_ * D_);
    f2bf_kernel<<<((long)D_ * D_ + 255) / 256, 256, 0, stream>>>(Wo, wob, (long)D_ * D_);

    // 2) history context chain (fp32, small)
    hist_embed_kernel<<<BS_ * TH_, 256, 0, stream>>>(hist, hm_W, hm_b, hm_g, hm_beta, ehb);
    ca_score_kernel<<<BS_ * TH_, 256, 0, stream>>>(ehb, ca_W1, ca_b1, ca_W2, ca_b2, awlb);
    ctx_pool_kernel<<<BS_, 256, 0, stream>>>(ehb, awlb, ctxb);
    kernel_gen_kernel<<<BS_, 256, 0, stream>>>(ctxb, kg_W1, kg_b1, kg_g, kg_beta,
                                               kg_W2, kg_b2, krnb);

    // 3) QKV projections: [4096,512] = x @ W.T  (bf16 out)
    {
        dim3 grid((D_ + 127) / 128, (int)(MBT / 16), 1);
        gemm_nt_bf16<<<grid, 256, 0, stream>>>(xb, wqb, nullptr, qb, nullptr,
            (int)MBT, D_, D_, D_, D_, D_, 0, 0, 0, 0, 0, 0, 1, 1.0f);
        gemm_nt_bf16<<<grid, 256, 0, stream>>>(xb, wkb, nullptr, kb, nullptr,
            (int)MBT, D_, D_, D_, D_, D_, 0, 0, 0, 0, 0, 0, 1, 1.0f);
        gemm_nt_bf16<<<grid, 256, 0, stream>>>(xb, wvb, nullptr, vb, nullptr,
            (int)MBT, D_, D_, D_, D_, D_, 0, 0, 0, 0, 0, 0, 1, 1.0f);
    }

    // 4) scores = Q·K^T / 8 per (bs,h): 32 x [1024,1024,64]  (fp32 out)
    {
        dim3 grid(T_ / 128, T_ / 16, BS_ * H_);
        gemm_nt_bf16<<<grid, 256, 0, stream>>>(qb, kb, scb, nullptr, nullptr,
            T_, T_, HD_, D_, D_, T_,
            (long)T_ * D_, (long)HD_,          // A: bs stride, head stride
            (long)T_ * D_, (long)HD_,          // B
            (long)H_ * T_ * T_, (long)T_ * T_, // C
            H_, 0.125f);
    }

    // 5) depthwise conv + causal mask + row softmax -> bf16 attn
    {
        dim3 grid(T_, BS_ * H_);
        conv_mask_softmax_kernel<<<grid, 256, 0, stream>>>(scb, krnb, atb);
    }

    // 6) V transpose and attn @ V (32 x [1024,64,1024]); 128 thr = 4 waves = 64 cols
    transpose_v_kernel<<<(MBT * D_ + 255) / 256, 256, 0, stream>>>(vb, vtb);
    {
        dim3 grid(1, T_ / 16, BS_ * H_);
        gemm_nt_bf16<<<grid, 128, 0, stream>>>(atb, vtb, nullptr, aob, nullptr,
            T_, HD_, T_, T_, T_, HD_,
            (long)H_ * T_ * T_, (long)T_ * T_,
            (long)H_ * HD_ * T_, (long)HD_ * T_,
            (long)H_ * T_ * HD_, (long)T_ * HD_,
            H_, 1.0f);
    }

    // 7) merge heads, final projection + bias (fp32 out -> d_out)
    merge_heads_kernel<<<(MBT * D_ + 255) / 256, 256, 0, stream>>>(aob, aomb);
    {
        dim3 grid((D_ + 127) / 128, (int)(MBT / 16), 1);
        gemm_nt_bf16<<<grid, 256, 0, stream>>>(aomb, wob, (float*)d_out, nullptr, bo,
            (int)MBT, D_, D_, D_, D_, D_, 0, 0, 0, 0, 0, 0, 1, 1.0f);
    }
}
